// AttentionBlock_1228360646991
// MI455X (gfx1250) — matrix operand order repaired and checked by reading, not compiled, not run on hardware
//
#include <hip/hip_runtime.h>

// ---------------------------------------------------------------------------
// AttentionBlock for MI455X (gfx1250, wave32, WMMA).
// Split-bf16 (hi/lo) GEMMs on v_wmma_f32_16x16x32_bf16, fp32 softmax in LDS.
// ---------------------------------------------------------------------------

typedef __bf16 bf16_t;
typedef __attribute__((ext_vector_type(16))) __bf16 v16bf;
typedef __attribute__((ext_vector_type(8)))  __bf16 v8bf;
typedef __attribute__((ext_vector_type(8)))  float  v8f;

constexpr int BB = 4;      // batch
constexpr int CC = 256;    // channels
constexpr int NN = 4096;   // H*W
constexpr int TE = 1024;   // temb dim
constexpr int SROW = 4100; // padded LDS row stride (bank de-conflict)

static __device__ __forceinline__ v8f wmma_bf16(v16bf a, v16bf b, v8f c) {
  return __builtin_amdgcn_wmma_f32_16x16x32_bf16(false, a, false, b, (short)0, c,
                                                 false, false);
}

static __device__ __forceinline__ void split_bf(float x, bf16_t& h, bf16_t& l) {
  h = (bf16_t)x;
  l = (bf16_t)(x - (float)h);
}

// A-fragment (16x32 bf16, M=row, K): per ISA 7.12.2 — lane<16: elems0-7 ->
// K+0..7, elems8-15 -> K+16..23; lane>=16: +8 / +24. src row-major [row][K].
static __device__ __forceinline__ v16bf load_a16(const bf16_t* __restrict__ src,
                                                 int row, int kBase, int ld, int hf) {
  const bf16_t* p = src + (size_t)row * ld + kBase + hf * 8;
  v8bf lo = *(const v8bf*)p;
  v8bf hi = *(const v8bf*)(p + 16);
  return __builtin_shufflevector(lo, hi, 0, 1, 2, 3, 4, 5, 6, 7, 8, 9, 10, 11,
                                 12, 13, 14, 15);
}

// B-fragment (32x16 bf16, K x N=col): lane<16: elems0-7 -> K+0..7, elems8-15 ->
// K+16..23; lane>=16: +8 / +24, column col = lane%16. src row-major [col][K] (i.e. B^T).
static __device__ __forceinline__ v16bf load_b16(const bf16_t* __restrict__ src,
                                                 int col, int kBase, int ld, int hf) {
  const bf16_t* p = src + (size_t)col * ld + kBase + hf * 8;
  v8bf lo = *(const v8bf*)p;
  v8bf hi = *(const v8bf*)(p + 16);
  return __builtin_shufflevector(lo, hi, 0, 1, 2, 3, 4, 5, 6, 7, 8, 9, 10, 11,
                                 12, 13, 14, 15);
}

// ---------------------------------------------------------------------------
// Stage 0: t[b][o] = temb[b] . t_w[o] + t_b[o]   (tiny, 1M MAC total)
// ---------------------------------------------------------------------------
__global__ void k_temb(const float* __restrict__ temb, const float* __restrict__ t_w,
                       const float* __restrict__ t_b, float* __restrict__ t) {
  int idx = blockIdx.x * blockDim.x + threadIdx.x;
  if (idx >= BB * CC) return;
  int b = idx >> 8, o = idx & 255;
  const float* te = temb + b * TE;
  const float* w = t_w + o * TE;
  float s = 0.f;
  for (int e = 0; e < TE; ++e) s += te[e] * w[e];
  t[idx] = s + t_b[o];
}

// ---------------------------------------------------------------------------
// Stage 0b: weights -> bf16 hi/lo (row-major [o][c], A-matrix friendly)
// ---------------------------------------------------------------------------
__global__ void k_prep_w(const float* __restrict__ qw, const float* __restrict__ kw,
                         const float* __restrict__ vw, const float* __restrict__ pw,
                         bf16_t* qwh, bf16_t* qwl, bf16_t* kwh, bf16_t* kwl,
                         bf16_t* vwh, bf16_t* vwl, bf16_t* pwh, bf16_t* pwl) {
  int idx = blockIdx.x * 256 + threadIdx.x; // 65536
  bf16_t h, l;
  split_bf(qw[idx], h, l); qwh[idx] = h; qwl[idx] = l;
  split_bf(kw[idx], h, l); kwh[idx] = h; kwl[idx] = l;
  split_bf(vw[idx], h, l); vwh[idx] = h; vwl[idx] = l;
  split_bf(pw[idx], h, l); pwh[idx] = h; pwl[idx] = l;
}

// ---------------------------------------------------------------------------
// Stage 0c: x [B][C][N] f32 -> xt hi/lo [B][N][C] bf16 (transposed for B-frags)
// ---------------------------------------------------------------------------
__global__ void k_prep_x(const float* __restrict__ x, bf16_t* __restrict__ xth,
                         bf16_t* __restrict__ xtl) {
  size_t idx = (size_t)blockIdx.x * 256 + threadIdx.x; // 16M
  int n = (int)(idx & (NN - 1));
  int c = (int)((idx >> 12) & 255);
  int b = (int)(idx >> 20);
  bf16_t h, l;
  split_bf(x[idx], h, l);
  size_t o = ((size_t)b * NN + n) * CC + c;
  xth[o] = h;
  xtl[o] = l;
}

// ---------------------------------------------------------------------------
// Stage 1: QKV projections.  out[o][n] = sum_c W[o][c] * x[c][n] + bias
// A = W (hi/lo), B = x^T (hi/lo).  Grid (N/16, B), 8 waves/block; each wave
// owns two 16-wide o-tiles for all of q,k,v.  Q,K stored [B][N][C] (A-matrix
// layout for attention), V stored [B][C][N] (B-matrix layout for P*V).
// ---------------------------------------------------------------------------
static __device__ __forceinline__ v8f proj_acc(const bf16_t* __restrict__ wh,
                                               const bf16_t* __restrict__ wl,
                                               const v16bf* xbh, const v16bf* xbl,
                                               int o0, int lm, int hf) {
  v8f acc = {0.f, 0.f, 0.f, 0.f, 0.f, 0.f, 0.f, 0.f};
#pragma unroll
  for (int kc = 0; kc < 8; ++kc) {
    v16bf ah = load_a16(wh, o0 + lm, kc * 32, CC, hf);
    v16bf al = load_a16(wl, o0 + lm, kc * 32, CC, hf);
    acc = wmma_bf16(ah, xbh[kc], acc);
    acc = wmma_bf16(ah, xbl[kc], acc);
    acc = wmma_bf16(al, xbh[kc], acc);
  }
  return acc;
}

__global__ void __launch_bounds__(256) k_qkv(
    const bf16_t* __restrict__ xth, const bf16_t* __restrict__ xtl,
    const bf16_t* __restrict__ qwh, const bf16_t* __restrict__ qwl,
    const bf16_t* __restrict__ kwh, const bf16_t* __restrict__ kwl,
    const bf16_t* __restrict__ vwh, const bf16_t* __restrict__ vwl,
    const float* __restrict__ q_b, const float* __restrict__ k_b,
    const float* __restrict__ v_b, const float* __restrict__ t,
    bf16_t* __restrict__ qh, bf16_t* __restrict__ ql,
    bf16_t* __restrict__ kh, bf16_t* __restrict__ kl,
    bf16_t* __restrict__ vh, bf16_t* __restrict__ vl) {
  int n0 = blockIdx.x * 16;
  int b = blockIdx.y;
  int lane = threadIdx.x & 31, wave = threadIdx.x >> 5;
  int hf = lane >> 4, lm = lane & 15;

  const bf16_t* xh = xth + (size_t)b * NN * CC;
  const bf16_t* xl = xtl + (size_t)b * NN * CC;
  v16bf xbh[8], xbl[8];
#pragma unroll
  for (int kc = 0; kc < 8; ++kc) {
    xbh[kc] = load_b16(xh, n0 + lm, kc * 32, CC, hf);
    xbl[kc] = load_b16(xl, n0 + lm, kc * 32, CC, hf);
  }
  const float* tb = t + b * CC;

#pragma unroll
  for (int oo = 0; oo < 2; ++oo) {
    int o0 = wave * 32 + oo * 16;
    // ---- Q: bias q_b[o] + t[b][o], store [B][N][C]
    v8f acc = proj_acc(qwh, qwl, xbh, xbl, o0, lm, hf);
#pragma unroll
    for (int r = 0; r < 8; ++r) {
      int o = o0 + r + hf * 8;
      float val = acc[r] + q_b[o] + tb[o];
      bf16_t h, l;
      split_bf(val, h, l);
      size_t ad = ((size_t)b * NN + n0 + lm) * CC + o;
      qh[ad] = h; ql[ad] = l;
    }
    // ---- K: bias k_b[o] + t[b][o], store [B][N][C]
    acc = proj_acc(kwh, kwl, xbh, xbl, o0, lm, hf);
#pragma unroll
    for (int r = 0; r < 8; ++r) {
      int o = o0 + r + hf * 8;
      float val = acc[r] + k_b[o] + tb[o];
      bf16_t h, l;
      split_bf(val, h, l);
      size_t ad = ((size_t)b * NN + n0 + lm) * CC + o;
      kh[ad] = h; kl[ad] = l;
    }
    // ---- V: bias v_b[o], store [B][C][N]
    acc = proj_acc(vwh, vwl, xbh, xbl, o0, lm, hf);
#pragma unroll
    for (int r = 0; r < 8; ++r) {
      int o = o0 + r + hf * 8;
      float val = acc[r] + v_b[o];
      bf16_t h, l;
      split_bf(val, h, l);
      size_t ad = ((size_t)b * CC + o) * NN + n0 + lm;
      vh[ad] = h; vl[ad] = l;
    }
  }
}

// ---------------------------------------------------------------------------
// Stage 2: attention.  One workgroup per (batch, 16-row query block).
// Full 16 x 4096 score row-block lives in LDS (16*4100*4 = 256.25 KB of the
// 320 KB WGP LDS).  Phase1: S = Q K^T * 1/16 (split-bf16 WMMA).  Softmax
// in-LDS (1 wave per row, wave32 shfl reductions).  Phase2: hh = P V^T,
// P re-split hi/lo from LDS, output stored [B][N][C] bf16 hi/lo.
// ---------------------------------------------------------------------------
__global__ void __launch_bounds__(256) k_attn(
    const bf16_t* __restrict__ qh_, const bf16_t* __restrict__ ql_,
    const bf16_t* __restrict__ kh_, const bf16_t* __restrict__ kl_,
    const bf16_t* __restrict__ vh_, const bf16_t* __restrict__ vl_,
    bf16_t* __restrict__ hhh, bf16_t* __restrict__ hhl) {
  extern __shared__ float S[]; // [16][SROW]
  int i0 = blockIdx.x * 16;
  int b = blockIdx.y;
  int lane = threadIdx.x & 31, wave = threadIdx.x >> 5;
  int hf = lane >> 4, lm = lane & 15;

  const bf16_t* qh = qh_ + (size_t)b * NN * CC;
  const bf16_t* ql = ql_ + (size_t)b * NN * CC;
  const bf16_t* kh = kh_ + (size_t)b * NN * CC;
  const bf16_t* kl = kl_ + (size_t)b * NN * CC;

  // Q A-fragments resident in registers (K = 256 -> 8 frags, hi+lo).
  v16bf qah[8], qal[8];
#pragma unroll
  for (int kc = 0; kc < 8; ++kc) {
    qah[kc] = load_a16(qh, i0 + lm, kc * 32, CC, hf);
    qal[kc] = load_a16(ql, i0 + lm, kc * 32, CC, hf);
  }

  // Phase 1: S tiles (16x16 each), 32 j-tiles per wave.
  for (int jt = wave; jt < NN / 16; jt += 8) {
    int j0 = jt * 16;
    v8f acc = {0.f, 0.f, 0.f, 0.f, 0.f, 0.f, 0.f, 0.f};
#pragma unroll
    for (int kc = 0; kc < 8; ++kc) {
      v16bf kbh = load_b16(kh, j0 + lm, kc * 32, CC, hf);
      v16bf kbl = load_b16(kl, j0 + lm, kc * 32, CC, hf);
      acc = wmma_bf16(qah[kc], kbh, acc);
      acc = wmma_bf16(qah[kc], kbl, acc);
      acc = wmma_bf16(qal[kc], kbh, acc);
    }
#pragma unroll
    for (int r = 0; r < 8; ++r)
      S[(r + hf * 8) * SROW + j0 + lm] = acc[r] * 0.0625f; // * C^-0.5
  }
  __syncthreads();

  // Softmax: wave w owns rows 2w, 2w+1.
#pragma unroll
  for (int rr = 0; rr < 2; ++rr) {
    float* Sr = S + (wave * 2 + rr) * SROW;
    float mx = -3.402823466e38f;
    for (int j = lane; j < NN; j += 32) mx = fmaxf(mx, Sr[j]);
#pragma unroll
    for (int o = 16; o > 0; o >>= 1) mx = fmaxf(mx, __shfl_xor(mx, o, 32));
    float sum = 0.f;
    for (int j = lane; j < NN; j += 32) {
      float e = __expf(Sr[j] - mx);
      Sr[j] = e;
      sum += e;
    }
#pragma unroll
    for (int o = 16; o > 0; o >>= 1) sum += __shfl_xor(sum, o, 32);
    float inv = 1.f / sum;
    for (int j = lane; j < NN; j += 32) Sr[j] *= inv;
  }
  __syncthreads();

  // Phase 2: hh[i][c] = sum_j P[i][j] * V[c][j].  Wave owns 2 c-tiles.
  const bf16_t* vh = vh_ + (size_t)b * CC * NN;
  const bf16_t* vl = vl_ + (size_t)b * CC * NN;
  int c0a = wave * 32, c0b = wave * 32 + 16;
  v8f acc0 = {0.f, 0.f, 0.f, 0.f, 0.f, 0.f, 0.f, 0.f};
  v8f acc1 = {0.f, 0.f, 0.f, 0.f, 0.f, 0.f, 0.f, 0.f};
  for (int j0 = 0; j0 < NN; j0 += 32) {
    v16bf ph, pl;
#pragma unroll
    for (int e = 0; e < 16; ++e) {
      int K = ((e >> 3) << 4) + hf * 8 + (e & 7); // A-layout K index
      float p = S[lm * SROW + j0 + K];
      bf16_t h = (bf16_t)p;
      ph[e] = h;
      pl[e] = (bf16_t)(p - (float)h);
    }
    v16bf vbh = load_b16(vh, c0a + lm, j0, NN, hf);
    v16bf vbl = load_b16(vl, c0a + lm, j0, NN, hf);
    acc0 = wmma_bf16(ph, vbh, acc0);
    acc0 = wmma_bf16(ph, vbl, acc0);
    acc0 = wmma_bf16(pl, vbh, acc0);
    vbh = load_b16(vh, c0b + lm, j0, NN, hf);
    vbl = load_b16(vl, c0b + lm, j0, NN, hf);
    acc1 = wmma_bf16(ph, vbh, acc1);
    acc1 = wmma_bf16(ph, vbl, acc1);
    acc1 = wmma_bf16(pl, vbh, acc1);
  }
#pragma unroll
  for (int r = 0; r < 8; ++r) {
    int m = r + hf * 8;
    size_t base = ((size_t)b * NN + i0 + m) * CC;
    bf16_t h, l;
    split_bf(acc0[r], h, l);
    hhh[base + c0a + lm] = h;
    hhl[base + c0a + lm] = l;
    split_bf(acc1[r], h, l);
    hhh[base + c0b + lm] = h;
    hhl[base + c0b + lm] = l;
  }
}

// ---------------------------------------------------------------------------
// Stage 3: out = x + p_w @ hh + p_b.  A = p_w hi/lo, B = hh [B][N][C] hi/lo.
// ---------------------------------------------------------------------------
__global__ void __launch_bounds__(256) k_out(
    const bf16_t* __restrict__ hhh, const bf16_t* __restrict__ hhl,
    const bf16_t* __restrict__ pwh, const bf16_t* __restrict__ pwl,
    const float* __restrict__ p_b, const float* __restrict__ x,
    float* __restrict__ out) {
  int n0 = blockIdx.x * 16;
  int b = blockIdx.y;
  int lane = threadIdx.x & 31, wave = threadIdx.x >> 5;
  int hf = lane >> 4, lm = lane & 15;

  const bf16_t* hh = hhh + (size_t)b * NN * CC;
  const bf16_t* hl = hhl + (size_t)b * NN * CC;
  v16bf bh[8], bl[8];
#pragma unroll
  for (int kc = 0; kc < 8; ++kc) {
    bh[kc] = load_b16(hh, n0 + lm, kc * 32, CC, hf);
    bl[kc] = load_b16(hl, n0 + lm, kc * 32, CC, hf);
  }

#pragma unroll
  for (int oo = 0; oo < 2; ++oo) {
    int o0 = wave * 32 + oo * 16;
    v8f acc = proj_acc(pwh, pwl, bh, bl, o0, lm, hf);
#pragma unroll
    for (int r = 0; r < 8; ++r) {
      int o = o0 + r + hf * 8;
      size_t ad = ((size_t)b * CC + o) * NN + n0 + lm;
      out[ad] = x[ad] + p_b[o] + acc[r];
    }
  }
}

// ---------------------------------------------------------------------------
// Launch
// ---------------------------------------------------------------------------
extern "C" void kernel_launch(void* const* d_in, const int* in_sizes, int n_in,
                              void* d_out, int out_size, void* d_ws, size_t ws_size,
                              hipStream_t stream) {
  (void)in_sizes; (void)n_in; (void)out_size; (void)ws_size;
  const float* x    = (const float*)d_in[0];
  const float* temb = (const float*)d_in[1];
  const float* q_w  = (const float*)d_in[2];
  const float* q_b  = (const float*)d_in[3];
  const float* k_w  = (const float*)d_in[4];
  const float* k_b  = (const float*)d_in[5];
  const float* v_w  = (const float*)d_in[6];
  const float* v_b  = (const float*)d_in[7];
  const float* p_w  = (const float*)d_in[8];
  const float* p_b  = (const float*)d_in[9];
  const float* t_w  = (const float*)d_in[10];
  const float* t_b  = (const float*)d_in[11];
  float* out = (float*)d_out;

  char* ws = (char*)d_ws;
  size_t off = 0;
  auto take = [&](size_t bytes) -> void* {
    void* p = (void*)(ws + off);
    off += (bytes + 255) & ~(size_t)255;
    return p;
  };
  const size_t WSZ = (size_t)CC * CC * sizeof(bf16_t);      // 128 KB
  const size_t TSZ = (size_t)BB * NN * CC * sizeof(bf16_t); // 8 MB

  float*  t   = (float*)take((size_t)BB * CC * sizeof(float));
  bf16_t* qwh = (bf16_t*)take(WSZ); bf16_t* qwl = (bf16_t*)take(WSZ);
  bf16_t* kwh = (bf16_t*)take(WSZ); bf16_t* kwl = (bf16_t*)take(WSZ);
  bf16_t* vwh = (bf16_t*)take(WSZ); bf16_t* vwl = (bf16_t*)take(WSZ);
  bf16_t* pwh = (bf16_t*)take(WSZ); bf16_t* pwl = (bf16_t*)take(WSZ);
  bf16_t* xth = (bf16_t*)take(TSZ); bf16_t* xtl = (bf16_t*)take(TSZ);
  bf16_t* qh  = (bf16_t*)take(TSZ); bf16_t* ql  = (bf16_t*)take(TSZ);
  bf16_t* kh  = (bf16_t*)take(TSZ); bf16_t* kl  = (bf16_t*)take(TSZ);
  bf16_t* vh  = (bf16_t*)take(TSZ); bf16_t* vl  = (bf16_t*)take(TSZ);
  bf16_t* hhh = (bf16_t*)take(TSZ); bf16_t* hhl = (bf16_t*)take(TSZ);

  k_temb<<<4, 256, 0, stream>>>(temb, t_w, t_b, t);
  k_prep_w<<<256, 256, 0, stream>>>(q_w, k_w, v_w, p_w, qwh, qwl, kwh, kwl,
                                    vwh, vwl, pwh, pwl);
  k_prep_x<<<65536, 256, 0, stream>>>(x, xth, xtl);
  k_qkv<<<dim3(NN / 16, BB), 256, 0, stream>>>(xth, xtl, qwh, qwl, kwh, kwl,
                                               vwh, vwl, q_b, k_b, v_b, t,
                                               qh, ql, kh, kl, vh, vl);
  k_attn<<<dim3(NN / 16, BB), 256, 16 * SROW * sizeof(float), stream>>>(
      qh, ql, kh, kl, vh, vl, hhh, hhl);
  k_out<<<dim3(NN / 16, BB), 256, 0, stream>>>(hhh, hhl, pwh, pwl, p_b, x, out);
}
